// AttentionHead_66718021976660
// MI455X (gfx1250) — matrix operand order repaired and checked by reading, not compiled, not run on hardware
//
#include <hip/hip_runtime.h>

typedef __attribute__((ext_vector_type(16))) __bf16 v16bf;
typedef __attribute__((ext_vector_type(8)))  __bf16 v8bf;
typedef __attribute__((ext_vector_type(8)))  float  v8f;

#define B_DIM 4
#define T_DIM 4096
#define C_DIM 768
#define H_DIM 64
#define NWT   192          // fused QKV output columns (3*64)

// ---- workspace layout (bytes) ----
#define WT_OFF  0u                       // Wt bf16 [192][768]  = 294912 B
#define QB_OFF  (512u * 1024u)           // Q bf16 [B*T][64]    = 2 MiB
#define KB_OFF  (QB_OFF + 2097152u)      // K bf16 [B*T][64]    = 2 MiB
#define VT_OFF  (KB_OFF + 2097152u)      // V^T bf16 [B][64][T] = 2 MiB

static __device__ __forceinline__ v16bf ld_pair(const __bf16* p0, const __bf16* p1) {
  v8bf a = *(const v8bf*)p0;
  v8bf b = *(const v8bf*)p1;
  return __builtin_shufflevector(a, b, 0,1,2,3,4,5,6,7,8,9,10,11,12,13,14,15);
}

static __device__ __forceinline__ v8f wmma_bf16(v16bf a, v16bf b, v8f c) {
  // D = A(16x32 bf16) * B(32x16 bf16) + C(16x16 f32)
  return __builtin_amdgcn_wmma_f32_16x16x32_bf16(false, a, false, b, (short)0, c,
                                                 false, false);
}

static __device__ __forceinline__ v16bf cvt_a_frag(const float* xr, int kk, int lh) {
  // A layout: e<8 -> k = lh*8+e ; e>=8 -> k = 16+lh*8+(e-8)
  const float4 f0 = *(const float4*)(xr + kk + lh * 8);
  const float4 f1 = *(const float4*)(xr + kk + lh * 8 + 4);
  const float4 f2 = *(const float4*)(xr + kk + 16 + lh * 8);
  const float4 f3 = *(const float4*)(xr + kk + 16 + lh * 8 + 4);
  v16bf a;
  a[0]  = (__bf16)f0.x; a[1]  = (__bf16)f0.y; a[2]  = (__bf16)f0.z; a[3]  = (__bf16)f0.w;
  a[4]  = (__bf16)f1.x; a[5]  = (__bf16)f1.y; a[6]  = (__bf16)f1.z; a[7]  = (__bf16)f1.w;
  a[8]  = (__bf16)f2.x; a[9]  = (__bf16)f2.y; a[10] = (__bf16)f2.z; a[11] = (__bf16)f2.w;
  a[12] = (__bf16)f3.x; a[13] = (__bf16)f3.y; a[14] = (__bf16)f3.z; a[15] = (__bf16)f3.w;
  return a;
}

// ------------------------------------------------------------------
// Kernel 0: fuse+transpose weights to bf16: Wt[n][c], n = {Q:0-63, K:64-127, V:128-191}
// ------------------------------------------------------------------
__global__ void wcvt_kernel(const float* __restrict__ Wq, const float* __restrict__ Wk,
                            const float* __restrict__ Wv, __bf16* __restrict__ Wt) {
  int idx = blockIdx.x * 256 + threadIdx.x;
  if (idx >= NWT * C_DIM) return;
  int n = idx / C_DIM, c = idx - n * C_DIM;
  int which = n >> 6, h = n & 63;
  const float* W = (which == 0) ? Wq : (which == 1) ? Wk : Wv;
  Wt[idx] = (__bf16)W[c * H_DIM + h];
}

// ------------------------------------------------------------------
// Kernel 1: fused QKV projection (bf16 WMMA), V written transposed.
// No LDS, no barriers: A fragments built directly from global fp32
// (each x element consumed by exactly one lane). grid = (B*T)/32,
// 128 threads; wave w -> col tile w of Q,K,V over 2 row tiles.
// ------------------------------------------------------------------
__global__ __launch_bounds__(128) void qkv_proj_kernel(
    const float* __restrict__ x, const __bf16* __restrict__ Wt,
    __bf16* __restrict__ Qb, __bf16* __restrict__ Kb, __bf16* __restrict__ Vt) {
  const int tid  = threadIdx.x;
  const int wave = tid >> 5;
  const int lane = tid & 31;
  const int lh   = lane >> 4;
  const int lm   = lane & 15;
  const int m0   = blockIdx.x * 32;              // global row (b*T + t)

  const float*  xr0 = x + (size_t)(m0 + lm) * C_DIM;
  const float*  xr1 = x + (size_t)(m0 + 16 + lm) * C_DIM;
  const __bf16* wq  = Wt + (size_t)(      wave * 16 + lm) * C_DIM;
  const __bf16* wk  = Wt + (size_t)( 64 + wave * 16 + lm) * C_DIM;
  const __bf16* wv  = Wt + (size_t)(128 + wave * 16 + lm) * C_DIM;

  v8f accQ[2] = {{}, {}}, accK[2] = {{}, {}}, accV[2] = {{}, {}};

  for (int kk = 0; kk < C_DIM; kk += 32) {
    const v16bf a0 = cvt_a_frag(xr0, kk, lh);
    const v16bf a1 = cvt_a_frag(xr1, kk, lh);
    // B fragments: n = Wt row, e<8 -> k = kk + lh*8 + e ; e>=8 -> k = kk + 16 + lh*8 + (e-8)
    const v16bf bq = ld_pair(wq + kk + lh * 8, wq + kk + 16 + lh * 8);
    const v16bf bk = ld_pair(wk + kk + lh * 8, wk + kk + 16 + lh * 8);
    const v16bf bv = ld_pair(wv + kk + lh * 8, wv + kk + 16 + lh * 8);
    accQ[0] = wmma_bf16(a0, bq, accQ[0]);  accQ[1] = wmma_bf16(a1, bq, accQ[1]);
    accK[0] = wmma_bf16(a0, bk, accK[0]);  accK[1] = wmma_bf16(a1, bk, accK[1]);
    accV[0] = wmma_bf16(a0, bv, accV[0]);  accV[1] = wmma_bf16(a1, bv, accV[1]);
  }

  // write-out: C/D layout m = r + 8*lh, n = lm
  const int b  = m0 / T_DIM;
  const int t0 = m0 - b * T_DIM;
#pragma unroll
  for (int mt = 0; mt < 2; ++mt) {
#pragma unroll
    for (int r = 0; r < 8; ++r) {
      const int m = mt * 16 + r + 8 * lh;
      Qb[(size_t)(m0 + m) * H_DIM + wave * 16 + lm] = (__bf16)accQ[mt][r];
      Kb[(size_t)(m0 + m) * H_DIM + wave * 16 + lm] = (__bf16)accK[mt][r];
      Vt[((size_t)b * H_DIM + wave * 16 + lm) * T_DIM + t0 + m] = (__bf16)accV[mt][r];
    }
  }
}

// ------------------------------------------------------------------
// Kernel 2: causal flash attention, work-balanced pairing + pipelined
// K/V staging. grid = B*32; block p handles q-tiles {p, 63-p} so every
// block runs exactly 65 key-iterations. 128 threads (4 waves x 16 rows).
// ------------------------------------------------------------------
__global__ __launch_bounds__(128) void attn_kernel(
    const __bf16* __restrict__ Qb, const __bf16* __restrict__ Kb,
    const __bf16* __restrict__ Vt, float* __restrict__ out) {
  __shared__ __bf16 Ks[64 * 80];        // [key][h]    stride 80
  __shared__ __bf16 Vs[64 * 80];        // [h][key]    stride 80
  __shared__ __bf16 Ps[4][16 * 80];     // per-wave P  [m][key] stride 80

  const int tid  = threadIdx.x;
  const int wave = tid >> 5;
  const int lane = tid & 31;
  const int lh   = lane >> 4;
  const int lm   = lane & 15;
  const int b    = blockIdx.x >> 5;
  const int pair = blockIdx.x & 31;
  const float scale = 0.03608439182435161f;   // 1/sqrt(C)

  const int srow = tid >> 1;               // staging: row 0..63
  const int sch  = (tid & 1) * 32;         // staging: col chunk 0/32
  const __bf16* kbase = Kb + (size_t)(b * T_DIM + srow) * H_DIM + sch;
  const __bf16* vbase = Vt + ((size_t)b * H_DIM + srow) * T_DIM + sch;

  for (int pass = 0; pass < 2; ++pass) {
    const int qb = pass ? (63 - pair) : pair;
    const int q0 = qb * 64 + wave * 16;

    // Q A-fragments in registers for the whole pass (k-dim = H = 2 steps of 32)
    v16bf qf[2];
    {
      const __bf16* qp = Qb + (size_t)(b * T_DIM + q0 + lm) * H_DIM;
#pragma unroll
      for (int ks = 0; ks < 2; ++ks)
        qf[ks] = ld_pair(qp + ks * 32 + lh * 8, qp + ks * 32 + 16 + lh * 8);
    }

    float mrow[8], lrow[8];
    v8f o[4] = {{}, {}, {}, {}};
#pragma unroll
    for (int r = 0; r < 8; ++r) { mrow[r] = -1e30f; lrow[r] = 0.0f; }

    // software pipeline: K/V tile for kb+1 loaded into registers while
    // WMMAs for kb run; prefetch kb+2 into caches.
    uint4 kreg[4], vreg[4];
    auto issue = [&](int kb) {
      const uint4* gk = (const uint4*)(kbase + (size_t)kb * 64 * H_DIM);
      kreg[0] = gk[0]; kreg[1] = gk[1]; kreg[2] = gk[2]; kreg[3] = gk[3];
      const uint4* gv = (const uint4*)(vbase + kb * 64);
      vreg[0] = gv[0]; vreg[1] = gv[1]; vreg[2] = gv[2]; vreg[3] = gv[3];
    };
    issue(0);

    for (int kb = 0; kb <= qb; ++kb) {
      { // drain staged registers into LDS (b128 stores)
        uint4* dk = (uint4*)&Ks[srow * 80 + sch];
        dk[0] = kreg[0]; dk[1] = kreg[1]; dk[2] = kreg[2]; dk[3] = kreg[3];
        uint4* dv = (uint4*)&Vs[srow * 80 + sch];
        dv[0] = vreg[0]; dv[1] = vreg[1]; dv[2] = vreg[2]; dv[3] = vreg[3];
      }
      if (kb < qb) issue(kb + 1);
      if (kb + 2 <= qb) {
        __builtin_prefetch(kbase + (size_t)(kb + 2) * 64 * H_DIM, 0, 0);
        __builtin_prefetch(vbase + (kb + 2) * 64, 0, 0);
      }
      __syncthreads();

      // ---- S = Q * K^T : 4 key tiles of 16, each 2 WMMA k-steps ----
      v8f s[4];
#pragma unroll
      for (int j = 0; j < 4; ++j) {
        v8f acc = {};
#pragma unroll
        for (int ks = 0; ks < 2; ++ks) {
          const __bf16* kp = &Ks[(j * 16 + lm) * 80 + ks * 32 + lh * 8];
          acc = wmma_bf16(qf[ks], ld_pair(kp, kp + 16), acc);
        }
        s[j] = acc;
      }

      // ---- scale + causal mask (diagonal block only) ----
      const bool diag = (kb == qb);
#pragma unroll
      for (int j = 0; j < 4; ++j) {
#pragma unroll
        for (int r = 0; r < 8; ++r) {
          float v = s[j][r] * scale;
          if (diag) {
            const int key = kb * 64 + j * 16 + lm;
            const int q   = q0 + r + 8 * lh;
            v = (key <= q) ? v : -1e30f;
          }
          s[j][r] = v;
        }
      }

      // ---- online softmax (8 rows/lane; reduce across 16-lane half) ----
      float alpha[8];
#pragma unroll
      for (int r = 0; r < 8; ++r) {
        float vmax = fmaxf(fmaxf(s[0][r], s[1][r]), fmaxf(s[2][r], s[3][r]));
        vmax = fmaxf(vmax, __shfl_xor(vmax, 1, 32));
        vmax = fmaxf(vmax, __shfl_xor(vmax, 2, 32));
        vmax = fmaxf(vmax, __shfl_xor(vmax, 4, 32));
        vmax = fmaxf(vmax, __shfl_xor(vmax, 8, 32));
        const float mn = fmaxf(mrow[r], vmax);
        alpha[r] = __expf(mrow[r] - mn);
        mrow[r] = mn;
        float rs = 0.0f;
#pragma unroll
        for (int j = 0; j < 4; ++j) {
          const float p = __expf(s[j][r] - mn);
          s[j][r] = p;
          rs += p;
        }
        rs += __shfl_xor(rs, 1, 32);
        rs += __shfl_xor(rs, 2, 32);
        rs += __shfl_xor(rs, 4, 32);
        rs += __shfl_xor(rs, 8, 32);
        lrow[r] = lrow[r] * alpha[r] + rs;
      }

      // ---- rescale running O ----
#pragma unroll
      for (int ht = 0; ht < 4; ++ht)
#pragma unroll
        for (int r = 0; r < 8; ++r) o[ht][r] *= alpha[r];

      // ---- bounce P (C-layout) through per-wave LDS into A-layout ----
      __bf16* pw = Ps[wave];
#pragma unroll
      for (int j = 0; j < 4; ++j)
#pragma unroll
        for (int r = 0; r < 8; ++r)
          pw[(r + 8 * lh) * 80 + j * 16 + lm] = (__bf16)s[j][r];

      asm volatile("s_wait_dscnt 0" ::: "memory");

      v16bf pf[2];
#pragma unroll
      for (int ks = 0; ks < 2; ++ks)
        pf[ks] = ld_pair(&pw[lm * 80 + ks * 32 + lh * 8],
                         &pw[lm * 80 + ks * 32 + 16 + lh * 8]);

      // ---- O += P * V : 4 h tiles, 2 WMMA k-steps over 64 keys ----
#pragma unroll
      for (int ht = 0; ht < 4; ++ht) {
#pragma unroll
        for (int ks = 0; ks < 2; ++ks) {
          const __bf16* vp = &Vs[(ht * 16 + lm) * 80 + ks * 32 + lh * 8];
          o[ht] = wmma_bf16(pf[ks], ld_pair(vp, vp + 16), o[ht]);
        }
      }
      __syncthreads();
    }

    // ---- epilogue: O / l, fp32 out ----
#pragma unroll
    for (int r = 0; r < 8; ++r) {
      const float inv = 1.0f / lrow[r];
      float* op = out + (size_t)(b * T_DIM + q0 + r + 8 * lh) * H_DIM + lm;
#pragma unroll
      for (int ht = 0; ht < 4; ++ht) op[ht * 16] = o[ht][r] * inv;
    }
  }
}

// ------------------------------------------------------------------
extern "C" void kernel_launch(void* const* d_in, const int* in_sizes, int n_in,
                              void* d_out, int out_size, void* d_ws, size_t ws_size,
                              hipStream_t stream) {
  (void)in_sizes; (void)n_in; (void)out_size; (void)ws_size;
  const float* x  = (const float*)d_in[0];
  const float* Wq = (const float*)d_in[1];
  const float* Wk = (const float*)d_in[2];
  const float* Wv = (const float*)d_in[3];
  float* out = (float*)d_out;

  char* ws = (char*)d_ws;
  __bf16* Wt = (__bf16*)(ws + WT_OFF);
  __bf16* Qb = (__bf16*)(ws + QB_OFF);
  __bf16* Kb = (__bf16*)(ws + KB_OFF);
  __bf16* Vt = (__bf16*)(ws + VT_OFF);

  wcvt_kernel<<<(NWT * C_DIM + 255) / 256, 256, 0, stream>>>(Wq, Wk, Wv, Wt);
  qkv_proj_kernel<<<(B_DIM * T_DIM) / 32, 128, 0, stream>>>(x, Wt, Qb, Kb, Vt);
  attn_kernel<<<B_DIM * 32, 128, 0, stream>>>(Qb, Kb, Vt, out);
}
